// StyleConditioningPipeline_33122787787409
// MI455X (gfx1250) — matrix operand order repaired and checked by reading, not compiled, not run on hardware
//
#include <hip/hip_runtime.h>
#include <hip/hip_bf16.h>
#include <math.h>

// ---------------------------------------------------------------------------
// StyleConditioningPipeline for MI455X (gfx1250, wave32, WMMA).
//
// Attention is degenerate (one kv token per batch -> softmax == 1), so the
// attn output per batch is (V[b]@wv+bv)@wo+bo broadcast over rows; wq/wk dead.
// Heavy work = two FFNs (~155 GFLOP) on v_wmma_f32_16x16x32_bf16.
// GEMM uses CDNA5 async global->LDS copies (ASYNCcnt) with double buffering.
// ---------------------------------------------------------------------------

typedef __attribute__((ext_vector_type(16))) __bf16 v16bf;
typedef __attribute__((ext_vector_type(8)))  float  v8f;

#define BB   16
#define TT   256
#define DDIM 512
#define FF   2048
#define D4   2048   // 4*D

__device__ inline void async_load_b128(unsigned lds_off, const __bf16* gaddr)
{
  // GLOBAL_LOAD_ASYNC_TO_LDS_B128: vdst = LDS byte address, vaddr = 64-bit
  // global address (GV mode). Tracked by ASYNCcnt.
  asm volatile("global_load_async_to_lds_b128 %0, %1, off"
               :: "v"(lds_off), "v"(gaddr)
               : "memory");
}

__device__ inline void wait_asynccnt0()
{
  asm volatile("s_wait_asynccnt 0x0" ::: "memory");
}

// ------------------------- WMMA GEMM ---------------------------------------
// C(MxN) = A(MxK) * B(KxN), B supplied transposed (Bt: N rows of K bf16).
// Block tile 128x128, 8 waves, each wave 32(M)x64(N) = 2x4 wmma tiles.
// K staged in 64-wide slabs, async-copied to LDS, double buffered.
// EPI==0: out_bf16 = bf16(gelu(acc + bias[n]))        (FFN layer 1)
// EPI==1: out_f32  = acc + bias[n] + resid[m,n]       (FFN layer 2, pre-LN)

union FragU { uint4 u[2]; v16bf v; };

template<int EPI>
__global__ __launch_bounds__(256)
void gemm_bf16_wmma(const __bf16* __restrict__ A, const __bf16* __restrict__ Bt,
                    const float* __restrict__ bias, const float* __restrict__ resid,
                    float* __restrict__ outF, __bf16* __restrict__ outB,
                    int M, int N, int Ktot)
{
  constexpr int LDSTR = 72;                     // 64 + 8 bf16 pad (bank spread)
  __shared__ __align__(16) __bf16 As[2][128 * LDSTR];
  __shared__ __align__(16) __bf16 Bs[2][128 * LDSTR];

  const int tid  = threadIdx.x;
  const int lane = tid & 31;
  const int wave = tid >> 5;
  const int wm   = wave & 3;     // M strip (4 strips of 32 rows)
  const int wn   = wave >> 2;    // N strip (2 strips of 64 cols)
  const int lm   = lane & 15;
  const int hx   = lane >> 4;    // lane half (ISA A/B fragment layout)

  const int m0 = blockIdx.y * 128;
  const int n0 = blockIdx.x * 128;

  // Staging coordinates: each thread owns col chunk scol of rows srow+32p.
  const int srow = tid >> 3;            // 0..31
  const int scol = (tid & 7) * 8;       // bf16 column within the 64-wide slab
  const __bf16* gA = A  + (size_t)(m0 + srow) * Ktot + scol;
  const __bf16* gB = Bt + (size_t)(n0 + srow) * Ktot + scol;
  // Low 32 bits of a generic LDS pointer == LDS byte offset (aperture rule).
  const unsigned ldsA[2] = { (unsigned)(uintptr_t)&As[0][srow * LDSTR + scol],
                             (unsigned)(uintptr_t)&As[1][srow * LDSTR + scol] };
  const unsigned ldsB[2] = { (unsigned)(uintptr_t)&Bs[0][srow * LDSTR + scol],
                             (unsigned)(uintptr_t)&Bs[1][srow * LDSTR + scol] };

  auto issue_slab = [&](int buf, int k0) {
#pragma unroll
    for (int p = 0; p < 4; ++p) {
      async_load_b128(ldsA[buf] + p * 32 * LDSTR * 2,
                      gA + (size_t)p * 32 * Ktot + k0);
      async_load_b128(ldsB[buf] + p * 32 * LDSTR * 2,
                      gB + (size_t)p * 32 * Ktot + k0);
    }
  };

  v8f acc[2][4] = {};

  issue_slab(0, 0);
  const int nslab = Ktot >> 6;
  for (int s = 0; s < nslab; ++s) {
    wait_asynccnt0();      // this wave's slab-s async copies have landed in LDS
    __syncthreads();       // all waves landed; all waves done with buf (s+1)&1
    if (s + 1 < nslab) issue_slab((s + 1) & 1, (s + 1) << 6);

    const __bf16* asb = As[s & 1];
    const __bf16* bsb = Bs[s & 1];
#pragma unroll
    for (int kk = 0; kk < 64; kk += 32) {
      FragU fa[2], fb[4];
      // A 16x32 bf16 fragment: lanes0-15 row M=lm, K=kb..kb+7 then kb+16..+23
#pragma unroll
      for (int i = 0; i < 2; ++i) {
        int r  = wm * 32 + i * 16 + lm;
        int kb = kk + (hx ? 8 : 0);
        fa[i].u[0] = *(const uint4*)(&asb[r * LDSTR + kb]);
        fa[i].u[1] = *(const uint4*)(&asb[r * LDSTR + kb + 16]);
      }
      // B 32x16 bf16 fragment (from N-major Bt): lane col N=lm, K=kb..kb+7 then kb+16..+23
#pragma unroll
      for (int j = 0; j < 4; ++j) {
        int r  = wn * 64 + j * 16 + lm;
        int kb = kk + (hx ? 8 : 0);
        fb[j].u[0] = *(const uint4*)(&bsb[r * LDSTR + kb]);
        fb[j].u[1] = *(const uint4*)(&bsb[r * LDSTR + kb + 16]);
      }
#pragma unroll
      for (int i = 0; i < 2; ++i)
#pragma unroll
        for (int j = 0; j < 4; ++j)
          acc[i][j] = __builtin_amdgcn_wmma_f32_16x16x32_bf16(
              false, fa[i].v, false, fb[j].v, (short)0, acc[i][j], false, false);
    }
  }

  // Epilogue. D layout: VGPR r -> M = r + 8*hx (per lane half), N = lm.
#pragma unroll
  for (int i = 0; i < 2; ++i) {
#pragma unroll
    for (int j = 0; j < 4; ++j) {
      int   ng = n0 + wn * 64 + j * 16 + lm;
      float bv = bias[ng];
#pragma unroll
      for (int r = 0; r < 8; ++r) {
        int    mg  = m0 + wm * 32 + i * 16 + r + hx * 8;
        size_t idx = (size_t)mg * N + ng;
        float  v   = acc[i][j][r] + bv;
        if (EPI == 0) {
          float g = 0.5f * v * (1.0f + erff(v * 0.70710678118654752f));
          outB[idx] = (__bf16)g;
        } else {
          outF[idx] = v + resid[idx];
        }
      }
    }
  }
}

// ---------------------- small helper kernels --------------------------------

// out[n*K + k] = bf16(w[k*N + n])  : f32 row-major (KxN) -> bf16 N-major (NxK)
__global__ void k_convert_wT(const float* __restrict__ w, __bf16* __restrict__ out,
                             int K, int N)
{
  size_t i = (size_t)blockIdx.x * blockDim.x + threadIdx.x;
  int n = (int)(i / K);
  int k = (int)(i % K);
  out[i] = (__bf16)w[(size_t)k * N + n];
}

__device__ inline void block_ln_stats(const float* rowS, float* redS, float* redQ,
                                      int tid, float& mean, float& inv)
{
  float a = rowS[tid], b = rowS[tid + 256];
  redS[tid] = a + b;
  redQ[tid] = a * a + b * b;
  __syncthreads();
  for (int o = 128; o > 0; o >>= 1) {
    if (tid < o) { redS[tid] += redS[tid + o]; redQ[tid] += redQ[tid + o]; }
    __syncthreads();
  }
  mean = redS[0] * (1.0f / DDIM);
  float var = redQ[0] * (1.0f / DDIM) - mean * mean;
  inv = rsqrtf(var + 1e-5f);
}

// K = LN(style@kproj_w + kproj_b), V likewise -> written straight into d_out.
__global__ __launch_bounds__(256)
void k_style_proj(const float* __restrict__ style,
                  const float* wK, const float* bK, const float* gK, const float* btK,
                  const float* wV, const float* bV, const float* gV, const float* btV,
                  float* __restrict__ outK, float* __restrict__ outV)
{
  int b = blockIdx.x, sel = blockIdx.y, tid = threadIdx.x;
  const float* w  = sel ? wV  : wK;
  const float* bi = sel ? bV  : bK;
  const float* g  = sel ? gV  : gK;
  const float* bt = sel ? btV : btK;
  float* out = sel ? outV : outK;

  __shared__ float rowS[DDIM];
  __shared__ float redS[256], redQ[256];
  const float* st = style + (size_t)b * DDIM;   // DS == D == 512
  for (int d = tid; d < DDIM; d += 256) {
    float acc = bi[d];
    for (int k = 0; k < DDIM; ++k) acc += st[k] * w[(size_t)k * DDIM + d];
    rowS[d] = acc;
  }
  __syncthreads();
  float mean, inv;
  block_ln_stats(rowS, redS, redQ, tid, mean, inv);
  for (int d = tid; d < DDIM; d += 256)
    out[(size_t)b * DDIM + d] = (rowS[d] - mean) * inv * g[d] + bt[d];
}

// o[L][b] = (V[b]@wv[L] + bv[L]) @ wo[L] + bo[L]   (the degenerate attention)
__global__ __launch_bounds__(256)
void k_attn_vec(const float* __restrict__ Vbuf,
                const float* __restrict__ wv, const float* __restrict__ bv,
                const float* __restrict__ wo, const float* __restrict__ bo,
                float* __restrict__ oBuf)
{
  int b = blockIdx.x, L = blockIdx.y, tid = threadIdx.x;
  const float* wvL = wv + (size_t)L * DDIM * DDIM;
  const float* woL = wo + (size_t)L * DDIM * DDIM;
  const float* bvL = bv + (size_t)L * DDIM;
  const float* boL = bo + (size_t)L * DDIM;
  __shared__ float uS[DDIM];
  const float* v = Vbuf + (size_t)b * DDIM;
  for (int d = tid; d < DDIM; d += 256) {
    float acc = bvL[d];
    for (int k = 0; k < DDIM; ++k) acc += v[k] * wvL[(size_t)k * DDIM + d];
    uS[d] = acc;
  }
  __syncthreads();
  for (int d = tid; d < DDIM; d += 256) {
    float acc = boL[d];
    for (int k = 0; k < DDIM; ++k) acc += uS[k] * woL[(size_t)k * DDIM + d];
    oBuf[((size_t)L * BB + b) * DDIM + d] = acc;
  }
}

// cumsum(durations) per batch, lengths -> d_out, searchsorted-right per frame.
__global__ __launch_bounds__(256)
void k_length_reg(const int* __restrict__ dur, int* __restrict__ tIdx,
                  int* __restrict__ validM, int* __restrict__ lenOut)
{
  int b = blockIdx.x, tid = threadIdx.x;
  __shared__ int cum[TT];
  int d = dur[(size_t)b * TT + tid];
  cum[tid] = d > 0 ? d : 0;
  __syncthreads();
  for (int o = 1; o < TT; o <<= 1) {
    int add = (tid >= o) ? cum[tid - o] : 0;
    __syncthreads();
    cum[tid] += add;
    __syncthreads();
  }
  int length = cum[TT - 1];
  if (tid == 0) lenOut[b] = length;
  for (int f = tid; f < FF; f += 256) {
    int lo = 0, hi = TT;
    while (lo < hi) { int mid = (lo + hi) >> 1; if (cum[mid] <= f) lo = mid + 1; else hi = mid; }
    int t = lo < TT - 1 ? lo : TT - 1;
    tIdx[(size_t)b * FF + f]   = t;
    validM[(size_t)b * FF + f] = (f < length) ? 1 : 0;
  }
}

// out = LN(x[row] + ovec[batch]); also emit bf16 copy for the GEMM A operand.
__global__ __launch_bounds__(256)
void k_add_ln(const float* __restrict__ x, const float* __restrict__ ovec,
              const float* __restrict__ g, const float* __restrict__ bt,
              float* __restrict__ outF, __bf16* __restrict__ outB, int rowsPerBatch)
{
  int row = blockIdx.x, tid = threadIdx.x;
  int b = row / rowsPerBatch;
  __shared__ float rowS[DDIM];
  __shared__ float redS[256], redQ[256];
  const float* xr = x + (size_t)row * DDIM;
  const float* ov = ovec + (size_t)b * DDIM;
  rowS[tid]       = xr[tid]       + ov[tid];
  rowS[tid + 256] = xr[tid + 256] + ov[tid + 256];
  __syncthreads();
  float mean, inv;
  block_ln_stats(rowS, redS, redQ, tid, mean, inv);
  for (int d = tid; d < DDIM; d += 256) {
    float y = (rowS[d] - mean) * inv * g[d] + bt[d];
    outF[(size_t)row * DDIM + d] = y;
    outB[(size_t)row * DDIM + d] = (__bf16)y;
  }
}

// upsampled row (gather or zero) + o1 -> LN -> {f32, bf16}
__global__ __launch_bounds__(256)
void k_gather_ln(const float* __restrict__ src, const int* __restrict__ tIdx,
                 const int* __restrict__ validM, const float* __restrict__ ovec,
                 const float* __restrict__ g, const float* __restrict__ bt,
                 float* __restrict__ outF, __bf16* __restrict__ outB)
{
  int row = blockIdx.x, tid = threadIdx.x;
  int b = row / FF;
  int t = tIdx[row];
  int valid = validM[row];
  __shared__ float rowS[DDIM];
  __shared__ float redS[256], redQ[256];
  const float* sr = src + ((size_t)b * TT + t) * DDIM;
  const float* ov = ovec + (size_t)b * DDIM;
  float s0 = valid ? sr[tid]       : 0.0f;
  float s1 = valid ? sr[tid + 256] : 0.0f;
  rowS[tid]       = s0 + ov[tid];
  rowS[tid + 256] = s1 + ov[tid + 256];
  __syncthreads();
  float mean, inv;
  block_ln_stats(rowS, redS, redQ, tid, mean, inv);
  for (int d = tid; d < DDIM; d += 256) {
    float y = (rowS[d] - mean) * inv * g[d] + bt[d];
    outF[(size_t)row * DDIM + d] = y;
    outB[(size_t)row * DDIM + d] = (__bf16)y;
  }
}

// plain row LN (in may alias out: row is fully buffered in LDS first)
__global__ __launch_bounds__(256)
void k_ln(const float* __restrict__ in, const float* __restrict__ g,
          const float* __restrict__ bt, float* __restrict__ out)
{
  int row = blockIdx.x, tid = threadIdx.x;
  __shared__ float rowS[DDIM];
  __shared__ float redS[256], redQ[256];
  const float* xr = in + (size_t)row * DDIM;
  rowS[tid]       = xr[tid];
  rowS[tid + 256] = xr[tid + 256];
  __syncthreads();
  float mean, inv;
  block_ln_stats(rowS, redS, redQ, tid, mean, inv);
  for (int d = tid; d < DDIM; d += 256)
    out[(size_t)row * DDIM + d] = (rowS[d] - mean) * inv * g[d] + bt[d];
}

// ---------------------------- launcher --------------------------------------

extern "C" void kernel_launch(void* const* d_in, const int* in_sizes, int n_in,
                              void* d_out, int out_size, void* d_ws, size_t ws_size,
                              hipStream_t stream)
{
  (void)in_sizes; (void)n_in; (void)out_size; (void)ws_size;

  const float* text    = (const float*)d_in[0];
  const float* style   = (const float*)d_in[1];
  const int*   dur     = (const int*)d_in[2];
  // d_in[3] = max_frame_len (compile-time FF); d_in[12/13/16/17] (wq,wk,bq,bk) dead.
  const float* kproj_w = (const float*)d_in[4];
  const float* kproj_b = (const float*)d_in[5];
  const float* kproj_g = (const float*)d_in[6];
  const float* kproj_bt= (const float*)d_in[7];
  const float* vproj_w = (const float*)d_in[8];
  const float* vproj_b = (const float*)d_in[9];
  const float* vproj_g = (const float*)d_in[10];
  const float* vproj_bt= (const float*)d_in[11];
  const float* attn_wv = (const float*)d_in[14];
  const float* attn_wo = (const float*)d_in[15];
  const float* attn_bv = (const float*)d_in[18];
  const float* attn_bo = (const float*)d_in[19];
  const float* ln1_g   = (const float*)d_in[20];
  const float* ln1_b   = (const float*)d_in[21];
  const float* ffn_w1  = (const float*)d_in[22];
  const float* ffn_b1  = (const float*)d_in[23];
  const float* ffn_w2  = (const float*)d_in[24];
  const float* ffn_b2  = (const float*)d_in[25];
  const float* ln2_g   = (const float*)d_in[26];
  const float* ln2_b   = (const float*)d_in[27];

  float* out = (float*)d_out;
  const size_t framesN = (size_t)BB * FF * DDIM;   // styled_frames
  float* outFrames = out;
  int*   outLen    = (int*)(out + framesN);
  float* outK      = out + framesN + BB;
  float* outV      = outK + (size_t)BB * DDIM;

  // workspace layout
  char* w = (char*)d_ws;
  auto alloc = [&](size_t bytes) -> char* {
    char* p = w; w += (bytes + 255) & ~(size_t)255; return p;
  };
  const size_t R0 = (size_t)BB * TT;   // 4096 rows
  const size_t R1 = (size_t)BB * FF;   // 32768 rows

  __bf16* w1T    = (__bf16*)alloc((size_t)2 * DDIM * D4 * 2);  // [2][N=2048][K=512]
  __bf16* w2T    = (__bf16*)alloc((size_t)2 * D4 * DDIM * 2);  // [2][N=512][K=2048]
  float*  oBuf   = (float*) alloc((size_t)2 * BB * DDIM * 4);  // attn vectors [2][B][D]
  int*    tIdx   = (int*)   alloc((size_t)BB * FF * 4);
  int*    validM = (int*)   alloc((size_t)BB * FF * 4);
  float*  styled = (float*) alloc(R0 * DDIM * 4);              // styled_text (persists)
  char*   arena  = w;                                          // phase-shared scratch
  // phase 0 (block 0)
  float*  xln0f = (float*) (arena);
  __bf16* xln0b = (__bf16*)(arena + R0 * DDIM * 4);
  __bf16* H0b   = (__bf16*)(arena + R0 * DDIM * 4 + R0 * DDIM * 2);
  float*  y0    = (float*) (arena + R0 * DDIM * 4 + R0 * DDIM * 2 + R0 * D4 * 2);
  // phase 1 (block 1) reuses the arena after phase 0 is consumed
  float*  xln1f = (float*) (arena);
  __bf16* xln1b = (__bf16*)(arena + R1 * DDIM * 4);
  __bf16* H1b   = (__bf16*)(arena + R1 * DDIM * 4 + R1 * DDIM * 2);

  // 1) weights -> bf16, transposed N-major (both layers)
  for (int L = 0; L < 2; ++L) {
    k_convert_wT<<<(DDIM * D4) / 256, 256, 0, stream>>>(
        ffn_w1 + (size_t)L * DDIM * D4, w1T + (size_t)L * D4 * DDIM, DDIM, D4);
    k_convert_wT<<<(D4 * DDIM) / 256, 256, 0, stream>>>(
        ffn_w2 + (size_t)L * D4 * DDIM, w2T + (size_t)L * DDIM * D4, D4, DDIM);
  }

  // 2) K, V -> d_out     3) degenerate attention vectors     4) length-regulate
  k_style_proj<<<dim3(BB, 2), 256, 0, stream>>>(
      style, kproj_w, kproj_b, kproj_g, kproj_bt,
             vproj_w, vproj_b, vproj_g, vproj_bt, outK, outV);
  k_attn_vec<<<dim3(BB, 2), 256, 0, stream>>>(outV, attn_wv, attn_bv, attn_wo,
                                              attn_bo, oBuf);
  k_length_reg<<<BB, 256, 0, stream>>>(dur, tIdx, validM, outLen);

  // ----- block 0 -----
  k_add_ln<<<(int)R0, 256, 0, stream>>>(text, oBuf, ln1_g, ln1_b, xln0f, xln0b, TT);
  gemm_bf16_wmma<0><<<dim3(D4 / 128, (int)R0 / 128), 256, 0, stream>>>(
      xln0b, w1T, ffn_b1, nullptr, nullptr, H0b, (int)R0, D4, DDIM);
  gemm_bf16_wmma<1><<<dim3(DDIM / 128, (int)R0 / 128), 256, 0, stream>>>(
      H0b, w2T, ffn_b2, xln0f, y0, nullptr, (int)R0, DDIM, D4);
  k_ln<<<(int)R0, 256, 0, stream>>>(y0, ln2_g, ln2_b, styled);

  // ----- length regulate + block 1 -----
  k_gather_ln<<<(int)R1, 256, 0, stream>>>(styled, tIdx, validM,
      oBuf + (size_t)BB * DDIM, ln1_g + DDIM, ln1_b + DDIM, xln1f, xln1b);
  gemm_bf16_wmma<0><<<dim3(D4 / 128, (int)R1 / 128), 256, 0, stream>>>(
      xln1b, w1T + (size_t)D4 * DDIM, ffn_b1 + D4, nullptr, nullptr, H1b,
      (int)R1, D4, DDIM);
  gemm_bf16_wmma<1><<<dim3(DDIM / 128, (int)R1 / 128), 256, 0, stream>>>(
      H1b, w2T + (size_t)DDIM * D4, ffn_b2 + DDIM, xln1f, outFrames, nullptr,
      (int)R1, DDIM, D4);
  k_ln<<<(int)R1, 256, 0, stream>>>(outFrames, ln2_g + DDIM, ln2_b + DDIM, outFrames);
}
